// MHSelfAttention_76897094468245
// MI455X (gfx1250) — matrix operand order repaired and checked by reading, not compiled, not run on hardware
//
#include <hip/hip_runtime.h>
#include <hip/hip_bf16.h>
#include <cstdint>
#include <cstddef>

// Problem dims (fixed by reference)
#define B_  2
#define S_  2048
#define D_  1024
#define H_  16
#define DK_ 64
#define N_QKV (3 * H_ * DK_)   // 3072
#define ROWS_ (B_ * S_)        // 4096

typedef __attribute__((ext_vector_type(16))) __bf16 v16bf;
typedef __attribute__((ext_vector_type(8)))  float  v8f;
typedef __attribute__((ext_vector_type(4)))  unsigned int v4u;

union FragAB { v16bf bf; v4u u[2]; };
union FragC  { v8f v; float f[8]; };

__device__ __forceinline__ unsigned short f2bf(float x) {
  unsigned int u = __float_as_uint(x);
  u += 0x7FFFu + ((u >> 16) & 1u);          // round-to-nearest-even
  return (unsigned short)(u >> 16);
}
__device__ __forceinline__ float bf2f(unsigned short h) {
  return __uint_as_float((unsigned int)h << 16);
}

// A-fragment (16x32, M x K): lane r=lane&15 holds row r.
// elements 0..7  = K[koff .. koff+7],  elements 8..15 = K[koff+16 .. koff+23],
// koff = (lane>>4)*8.   (ISA 7.12.2, 16-bit A matrix)
__device__ __forceinline__ FragAB load_a(const unsigned short* __restrict__ base, int ld) {
  const int lane = threadIdx.x & 31;
  const unsigned short* p = base + (size_t)(lane & 15) * ld + ((lane >> 4) << 3);
  FragAB f;
  f.u[0] = *reinterpret_cast<const v4u*>(p);
  f.u[1] = *reinterpret_cast<const v4u*>(p + 16);
  return f;
}

// B-fragment (32x16, K x N) loaded from Bt stored row-major as N x K:
// lane n=lane&15 holds column n; elements 0..7 = K[koff .. koff+7], 8..15 = K[koff+16 .. koff+23], koff = (lane>>4)*8.
__device__ __forceinline__ FragAB load_b(const unsigned short* __restrict__ base, int ld) {
  const int lane = threadIdx.x & 31;
  const unsigned short* p = base + (size_t)(lane & 15) * ld + ((lane >> 4) << 3);
  FragAB f;
  f.u[0] = *reinterpret_cast<const v4u*>(p);
  f.u[1] = *reinterpret_cast<const v4u*>(p + 16);
  return f;
}

__device__ __forceinline__ v8f wmma_bf16(const FragAB& a, const FragAB& b, v8f c) {
  return __builtin_amdgcn_wmma_f32_16x16x32_bf16(false, a.bf, false, b.bf,
                                                 (short)0, c, false, false);
}

// ---------------------------------------------------------------- convert
__global__ __launch_bounds__(256) void cvt_bf16_kernel(const float* __restrict__ src,
                                                       unsigned short* __restrict__ dst,
                                                       int n) {
  int i = (blockIdx.x * blockDim.x + threadIdx.x) * 4;
  if (i + 3 < n) {
    float4 v = *reinterpret_cast<const float4*>(src + i);
    unsigned int lo = (unsigned int)f2bf(v.x) | ((unsigned int)f2bf(v.y) << 16);
    unsigned int hi = (unsigned int)f2bf(v.z) | ((unsigned int)f2bf(v.w) << 16);
    uint2 pk; pk.x = lo; pk.y = hi;
    *reinterpret_cast<uint2*>(dst + i) = pk;
  }
}

// ---------------------------------------------------------------- QKV GEMM
// C[r,n] = sum_m xb[r,m] * wqkv[n,m];  r = b*S+s, n = (q*H+h)*DK+d
// 4 waves/block, each wave computes a 64x64 tile (4x4 WMMA fragments):
// 16 v_wmma per k-step vs 16 global_load_b128 -> 1:1 issue mix, 4x frag reuse.
// Epilogue scatters bf16 into Q[b,h,s,d], K[b,h,s,d], Vt[b,h,d,s].
__global__ __launch_bounds__(128) void qkv_gemm_kernel(
    const unsigned short* __restrict__ xb, const unsigned short* __restrict__ wb,
    unsigned short* __restrict__ Qb, unsigned short* __restrict__ Kb,
    unsigned short* __restrict__ Vt) {
  const int lane = threadIdx.x & 31;
  const int wid  = threadIdx.x >> 5;               // 4 waves, 2x2 wave grid
  const int rbase = blockIdx.x * 128 + (wid >> 1) * 64;
  const int cbase = blockIdx.y * 128 + (wid & 1) * 64;

  FragC acc[4][4];
  for (int i = 0; i < 4; ++i)
    for (int j = 0; j < 4; ++j)
      for (int e = 0; e < 8; ++e) acc[i][j].f[e] = 0.0f;

  for (int k = 0; k < D_; k += 32) {
    __builtin_prefetch(xb + (size_t)rbase * D_ + k + 128, 0, 3);
    FragAB a[4], b[4];
#pragma unroll
    for (int i = 0; i < 4; ++i)
      a[i] = load_a(xb + (size_t)(rbase + i * 16) * D_ + k, D_);
#pragma unroll
    for (int j = 0; j < 4; ++j)
      b[j] = load_b(wb + (size_t)(cbase + j * 16) * D_ + k, D_);
#pragma unroll
    for (int i = 0; i < 4; ++i)
#pragma unroll
      for (int j = 0; j < 4; ++j)
        acc[i][j].v = wmma_bf16(a[i], b[j], acc[i][j].v);
  }

  const int rh = (lane >> 4) * 8;
  const int cl = lane & 15;
  for (int i = 0; i < 4; ++i)
    for (int j = 0; j < 4; ++j)
      for (int e = 0; e < 8; ++e) {
        int row = rbase + i * 16 + rh + e;
        int col = cbase + j * 16 + cl;
        unsigned short bv = f2bf(acc[i][j].f[e]);
        int q = col >> 10, rem = col & 1023;
        int h = rem >> 6,  d   = rem & 63;
        int bb = row >> 11, s  = row & 2047;
        size_t bh = (size_t)(bb * H_ + h);
        if (q == 0)      Qb[(bh * S_ + s) * DK_ + d] = bv;
        else if (q == 1) Kb[(bh * S_ + s) * DK_ + d] = bv;
        else             Vt[(bh * DK_ + d) * S_ + s] = bv;
      }
}

// ---------------------------------------------------------------- RoPE (in place on Q,K)
__global__ __launch_bounds__(256) void rope_kernel(unsigned short* __restrict__ Qb,
                                                   unsigned short* __restrict__ Kb) {
  unsigned int idx = blockIdx.x * blockDim.x + threadIdx.x;   // 2^21 threads
  int i  = idx & 15;
  int s  = (idx >> 4) & 2047;
  int h  = (idx >> 15) & 15;
  int qk = (idx >> 19) & 1;
  int b  = (idx >> 20) & 1;
  unsigned short* buf = qk ? Kb : Qb;
  size_t base = (((size_t)(b * H_ + h)) * S_ + s) * DK_;
  float x1 = bf2f(buf[base + i]);
  float x2 = bf2f(buf[base + i + 16]);
  float invf = __powf(6000.0f, -(float)i * (1.0f / 16.0f));
  float ang = (float)s * invf;
  float sn, c;
  __sincosf(ang, &sn, &c);
  buf[base + i]      = f2bf(x1 * c - x2 * sn);
  buf[base + i + 16] = f2bf(x2 * c + x1 * sn);
}

// ---------------------------------------------------------------- flash attention
// One wave per (b,h,16-row q-tile). K processed in 32-column blocks.
__global__ __launch_bounds__(128) void attn_kernel(
    const unsigned short* __restrict__ Qb, const unsigned short* __restrict__ Kb,
    const unsigned short* __restrict__ Vt, unsigned short* __restrict__ attnb) {
  __shared__ __align__(16) unsigned short psh[4][16 * 40];   // 16x32 P tile, padded ld=40

  const int lane = threadIdx.x & 31;
  const int wid  = threadIdx.x >> 5;
  const int gw = blockIdx.x * 4 + wid;     // 4096 waves total
  const int qt = gw & 127;
  const int h  = (gw >> 7) & 15;
  const int b  = (gw >> 11) & 1;
  const int qbase = qt * 16;
  const size_t bh = (size_t)(b * H_ + h);
  const int rh = (lane >> 4) * 8;
  const int cl = lane & 15;

  const unsigned short* Qp = Qb + (bh * S_ + qbase) * DK_;
  FragAB qa0 = load_a(Qp, DK_);
  FragAB qa1 = load_a(Qp + 32, DK_);

  FragC o[4];
  float rmax[8], rsum[8];
  for (int e = 0; e < 8; ++e) { rmax[e] = -1e30f; rsum[e] = 0.0f; }
  for (int g = 0; g < 4; ++g)
    for (int e = 0; e < 8; ++e) o[g].f[e] = 0.0f;

  FragC zz;
  for (int e = 0; e < 8; ++e) zz.f[e] = 0.0f;

  for (int kb = 0; kb < qbase + 16; kb += 32) {
    const unsigned short* Kp = Kb + (bh * S_ + kb) * DK_;
    FragAB k00 = load_b(Kp, DK_);                 // cols kb..kb+15, d 0..31
    FragAB k01 = load_b(Kp + 32, DK_);            // cols kb..kb+15, d 32..63
    FragAB k10 = load_b(Kp + 16 * DK_, DK_);      // cols kb+16..kb+31
    FragAB k11 = load_b(Kp + 16 * DK_ + 32, DK_);
    FragC s0, s1;
    s0.v = wmma_bf16(qa0, k00, zz.v);
    s0.v = wmma_bf16(qa1, k01, s0.v);
    s1.v = wmma_bf16(qa0, k10, zz.v);
    s1.v = wmma_bf16(qa1, k11, s1.v);

    float tmax[8], alpha[8];
    for (int e = 0; e < 8; ++e) {
      int row = qbase + rh + e;
      float a  = s0.f[e] * 0.125f;                 // 1/sqrt(64)
      float c2 = s1.f[e] * 0.125f;
      if (kb + cl > row)      a  = -1e30f;         // causal mask
      if (kb + 16 + cl > row) c2 = -1e30f;
      s0.f[e] = a; s1.f[e] = c2;
      float m = fmaxf(a, c2);
      for (int msk = 1; msk < 16; msk <<= 1)
        m = fmaxf(m, __shfl_xor(m, msk, 32));      // row lives in one 16-lane half
      tmax[e] = m;
    }
    for (int e = 0; e < 8; ++e) {
      float nm = fmaxf(rmax[e], tmax[e]);
      alpha[e] = __expf(rmax[e] - nm);
      float p0 = __expf(s0.f[e] - nm);
      float p1 = __expf(s1.f[e] - nm);
      s0.f[e] = p0; s1.f[e] = p1;
      float t = p0 + p1;
      for (int msk = 1; msk < 16; msk <<= 1)
        t += __shfl_xor(t, msk, 32);
      rmax[e] = nm;
      rsum[e] = rsum[e] * alpha[e] + t;
    }
    for (int g = 0; g < 4; ++g)
      for (int e = 0; e < 8; ++e) o[g].f[e] *= alpha[e];

    // stage P (C layout -> row-major LDS) then reload as A-fragment
    unsigned short* pl = &psh[wid][0];
    for (int e = 0; e < 8; ++e) {
      int prow = rh + e;
      pl[prow * 40 + cl]      = f2bf(s0.f[e]);
      pl[prow * 40 + 16 + cl] = f2bf(s1.f[e]);
    }
    asm volatile("s_wait_dscnt 0" ::: "memory");
    FragAB pf = load_a(pl, 40);

    const unsigned short* Vp = Vt + (bh * DK_) * S_ + kb;
    for (int g = 0; g < 4; ++g) {
      FragAB vf = load_b(Vp + (size_t)(g * 16) * S_, S_);   // 32 k-vals contiguous per lane
      o[g].v = wmma_bf16(pf, vf, o[g].v);
    }
  }

  for (int g = 0; g < 4; ++g)
    for (int e = 0; e < 8; ++e) {
      int row = qbase + rh + e;
      int col = h * DK_ + g * 16 + cl;
      float ov = o[g].f[e] / rsum[e];
      attnb[((size_t)(b * S_ + row)) * (H_ * DK_) + col] = f2bf(ov);
    }
}

// ---------------------------------------------------------------- output GEMM
// out[r,n] = sum_k attnb[r,k] * Wo[n,k]; same 4x4-fragment wave tiling.
__global__ __launch_bounds__(128) void out_gemm_kernel(
    const unsigned short* __restrict__ ab, const unsigned short* __restrict__ wob,
    float* __restrict__ out) {
  const int lane = threadIdx.x & 31;
  const int wid  = threadIdx.x >> 5;
  const int rbase = blockIdx.x * 128 + (wid >> 1) * 64;
  const int cbase = blockIdx.y * 128 + (wid & 1) * 64;

  FragC acc[4][4];
  for (int i = 0; i < 4; ++i)
    for (int j = 0; j < 4; ++j)
      for (int e = 0; e < 8; ++e) acc[i][j].f[e] = 0.0f;

  for (int k = 0; k < D_; k += 32) {
    FragAB a[4], b[4];
#pragma unroll
    for (int i = 0; i < 4; ++i)
      a[i] = load_a(ab + (size_t)(rbase + i * 16) * D_ + k, D_);
#pragma unroll
    for (int j = 0; j < 4; ++j)
      b[j] = load_b(wob + (size_t)(cbase + j * 16) * D_ + k, D_);
#pragma unroll
    for (int i = 0; i < 4; ++i)
#pragma unroll
      for (int j = 0; j < 4; ++j)
        acc[i][j].v = wmma_bf16(a[i], b[j], acc[i][j].v);
  }

  const int rh = (lane >> 4) * 8;
  const int cl = lane & 15;
  for (int i = 0; i < 4; ++i)
    for (int j = 0; j < 4; ++j)
      for (int e = 0; e < 8; ++e) {
        int row = rbase + i * 16 + rh + e;
        int col = cbase + j * 16 + cl;
        out[(size_t)row * D_ + col] = acc[i][j].f[e];
      }
}

// ---------------------------------------------------------------- launch
extern "C" void kernel_launch(void* const* d_in, const int* in_sizes, int n_in,
                              void* d_out, int out_size, void* d_ws, size_t ws_size,
                              hipStream_t stream) {
  (void)in_sizes; (void)n_in; (void)out_size; (void)ws_size;
  const float* x    = (const float*)d_in[0];
  const float* Wqkv = (const float*)d_in[1];
  const float* Wo   = (const float*)d_in[2];
  float* out = (float*)d_out;

  char* ws = (char*)d_ws;
  size_t off = 0;
  auto carve = [&](size_t bytes) -> void* {
    void* p = ws + off;
    off += (bytes + 255) & ~(size_t)255;
    return p;
  };
  const size_t nQKV = (size_t)B_ * H_ * S_ * DK_;      // 4,194,304
  unsigned short* Qb    = (unsigned short*)carve(nQKV * 2);
  unsigned short* Kb    = (unsigned short*)carve(nQKV * 2);
  unsigned short* Vt    = (unsigned short*)carve(nQKV * 2);
  unsigned short* attnb = (unsigned short*)carve((size_t)ROWS_ * D_ * 2);
  unsigned short* xb    = (unsigned short*)carve((size_t)ROWS_ * D_ * 2);
  unsigned short* wqkvb = (unsigned short*)carve((size_t)N_QKV * D_ * 2);
  unsigned short* wob   = (unsigned short*)carve((size_t)D_ * D_ * 2);

  // 1) downconvert inputs to bf16
  int n1 = ROWS_ * D_;                      // 4,194,304
  int n2 = N_QKV * D_;                      // 3,145,728
  int n3 = D_ * H_ * DK_;                   // 1,048,576
  cvt_bf16_kernel<<<n1 / 1024, 256, 0, stream>>>(x, xb, n1);
  cvt_bf16_kernel<<<n2 / 1024, 256, 0, stream>>>(Wqkv, wqkvb, n2);
  cvt_bf16_kernel<<<n3 / 1024, 256, 0, stream>>>(Wo, wob, n3);

  // 2) QKV projection (M=4096, N=3072, K=1024), 128x128 block tiles
  qkv_gemm_kernel<<<dim3(ROWS_ / 128, N_QKV / 128), 128, 0, stream>>>(xb, wqkvb, Qb, Kb, Vt);

  // 3) RoPE on Q and K (first 32 dims)
  rope_kernel<<<(B_ * 2 * H_ * S_ * 16) / 256, 256, 0, stream>>>(Qb, Kb);

  // 4) causal flash attention, one wave per 16-row q-tile
  attn_kernel<<<(B_ * H_ * (S_ / 16)) / 4, 128, 0, stream>>>(Qb, Kb, Vt, attnb);

  // 5) output projection (M=4096, N=1024, K=1024), 128x128 block tiles
  out_gemm_kernel<<<dim3(ROWS_ / 128, D_ / 128), 128, 0, stream>>>(attnb, wob, out);
}
